// MSA_Block_10230612099916
// MI455X (gfx1250) — matrix operand order repaired and checked by reading, not compiled, not run on hardware
//
#include <hip/hip_runtime.h>
#include <hip/hip_bf16.h>

// ---------------------------------------------------------------------------
// MI455X / gfx1250 implementation of the 3D windowed-attention + ACmix block.
// GEMM-shaped stages use V_WMMA_F32_16X16X32_F16 (f16 in, f32 accumulate).
// Constant-table LDS staging uses GLOBAL_LOAD_ASYNC_TO_LDS_B128 (ASYNCcnt).
// ---------------------------------------------------------------------------

typedef __attribute__((ext_vector_type(16))) _Float16 v16h;
typedef __attribute__((ext_vector_type(2)))  __fp16   v2fp;
typedef __attribute__((ext_vector_type(8)))  float    v8f;

union Frag {
    v16h v;
    _Float16 h[16];
    unsigned u[8];
};

__device__ __forceinline__ v8f wmma_f32_f16(v16h a, v16h b, v8f c) {
    // 8 args: (neg_a, A, neg_b, B, c_mod, C, reuse_a, reuse_b)
    return __builtin_amdgcn_wmma_f32_16x16x32_f16(false, a, false, b,
                                                  (short)0, c, false, false);
}

// Pack two f32 into one dword of two f16 with a single v_cvt_pk_rtz_f16_f32.
__device__ __forceinline__ unsigned pk2(float x, float y) {
    union { v2fp v; unsigned u; } c;
    c.v = __builtin_amdgcn_cvt_pkrtz(x, y);
    return c.u;
}

// Async copy of 16 bytes global -> LDS (wave-relative LDS byte offset in VGPR).
__device__ __forceinline__ void async_g2l_b128(unsigned ldsoff, const void* gaddr) {
    asm volatile("global_load_async_to_lds_b128 %0, %1, off"
                 :: "v"(ldsoff), "v"(gaddr) : "memory");
}
__device__ __forceinline__ void wait_asynccnt0() {
    asm volatile("s_wait_asynccnt 0x0" ::: "memory");
}
// Low 32 bits of a generic LDS pointer == wave-relative LDS byte offset.
__device__ __forceinline__ unsigned lds_off(const void* p) {
    return (unsigned)(size_t)p;
}

// Problem constants
#define EMB   256
#define QKV_N 768
#define LVOX  16384          // 16*32*32 voxels per batch
#define HEADS 8
#define HD    32

// ---------------------------------------------------------------------------
// Kernel 1: qkv = x_in @ qkv_w        [65536,256] x [256,768] -> [65536,768]
// 128 threads (4 waves), tile 64x64, K-step 32, double-buffered LDS.
// B tile transposed in registers: 4x4 sub-blocks, K-pairs packed via
// v_cvt_pk_rtz_f16_f32, stored as single b64 per N column.
// ---------------------------------------------------------------------------
__global__ __launch_bounds__(128) void qkv_gemm(const float* __restrict__ A,
                                                const float* __restrict__ Bw,
                                                float* __restrict__ C) {
    __shared__ __attribute__((aligned(16))) _Float16 As[2][64][40];  // [buf][m][k]
    __shared__ __attribute__((aligned(16))) _Float16 Bs[2][64][40];  // [buf][n][k]

    const int tid  = threadIdx.x;
    const int lane = tid & 31;
    const int wave = tid >> 5;
    const int l15  = lane & 15;
    const int half = lane >> 4;

    const int tileN = blockIdx.x * 64;
    const int tileM = blockIdx.y * 64;
    const int waveM = (wave >> 1) * 32;
    const int waveN = (wave & 1) * 32;

    // A: thread loads 4 float4 (row-major). B: thread owns 4(K)x4(N) sub-block.
    const int bK = (tid >> 4) * 4;       // 0..28
    const int bN = (tid & 15) * 4;       // 0..60

    float4 ra[4], rb[4];

    auto loadA = [&](int k0) {
#pragma unroll
        for (int j = 0; j < 4; ++j) {
            int i = tid + j * 128;
            int row = (i * 4) >> 5, col = (i * 4) & 31;
            ra[j] = *(const float4*)(A + (size_t)(tileM + row) * EMB + k0 + col);
        }
    };
    auto loadB = [&](int k0) {
#pragma unroll
        for (int r = 0; r < 4; ++r)
            rb[r] = *(const float4*)(Bw + (size_t)(k0 + bK + r) * QKV_N + tileN + bN);
    };
    auto stage = [&](int buf) {
#pragma unroll
        for (int j = 0; j < 4; ++j) {
            int i = tid + j * 128;
            int row = (i * 4) >> 5, col = (i * 4) & 31;
            *(uint2*)&As[buf][row][col] =
                make_uint2(pk2(ra[j].x, ra[j].y), pk2(ra[j].z, ra[j].w));
        }
        const float* c0 = &rb[0].x;      // rb[r][nn] access
        const float* c1 = &rb[1].x;
        const float* c2 = &rb[2].x;
        const float* c3 = &rb[3].x;
#pragma unroll
        for (int nn = 0; nn < 4; ++nn) {
            *(uint2*)&Bs[buf][bN + nn][bK] =
                make_uint2(pk2(c0[nn], c1[nn]), pk2(c2[nn], c3[nn]));
        }
    };

    v8f acc[2][2] = {};

    loadA(0); loadB(0);
    stage(0);
    __syncthreads();

    for (int it = 0; it < 8; ++it) {
        const int buf = it & 1;
        if (it < 7) { loadA((it + 1) * 32); loadB((it + 1) * 32); }

        Frag a[2], b[2];
#pragma unroll
        for (int mb = 0; mb < 2; ++mb) {
            int row = waveM + mb * 16 + l15;
#pragma unroll
            for (int vg = 0; vg < 8; ++vg) {
                int kk = 2 * vg + 16 * half;
                a[mb].u[vg] = *(const unsigned*)&As[buf][row][kk];
            }
        }
#pragma unroll
        for (int nb = 0; nb < 2; ++nb) {
            int n = waveN + nb * 16 + l15;
#pragma unroll
            for (int vg = 0; vg < 8; ++vg) {
                int kk = 2 * vg + 16 * half;
                b[nb].u[vg] = *(const unsigned*)&Bs[buf][n][kk];
            }
        }
#pragma unroll
        for (int mb = 0; mb < 2; ++mb)
#pragma unroll
            for (int nb = 0; nb < 2; ++nb)
                acc[mb][nb] = wmma_f32_f16(a[mb].v, b[nb].v, acc[mb][nb]);

        if (it < 7) {
            stage(1 - buf);
            __syncthreads();
        }
    }

#pragma unroll
    for (int mb = 0; mb < 2; ++mb)
#pragma unroll
        for (int nb = 0; nb < 2; ++nb)
#pragma unroll
            for (int r = 0; r < 8; ++r) {
                int row = tileM + waveM + mb * 16 + r + 8 * half;
                int col = tileN + waveN + nb * 16 + l15;
                C[(size_t)row * QKV_N + col] = acc[mb][nb][r];
            }
}

// Token -> flat voxel position (b,z,y,x packed over D=16,H=32,W=32)
__device__ __forceinline__ int tok_pos(int b, int zb, int yb, int xb, int t) {
    int tz = t >> 4, ty = (t >> 2) & 3, tx = t & 3;
    return ((b * 16 + zb * 4 + tz) * 32 + yb * 4 + ty) * 32 + xb * 4 + tx;
}

// ---------------------------------------------------------------------------
// Kernel 2: windowed attention. grid (2, 1024), 128 threads (4 waves).
// wave = one head of one window. S=Q.K^T (16 WMMA), softmax+bias, O=P.V (16).
// ---------------------------------------------------------------------------
__global__ __launch_bounds__(128) void attn_kernel(const float* __restrict__ qkv,
                                                   const float* __restrict__ rpb,
                                                   float* __restrict__ attn_out) {
    __shared__ float    rpbs[343 * HEADS];      // 10976 B (686 x b128)
    __shared__ _Float16 P[4][64][72];           // 36864 B, per-wave prob tile

    const int tid = threadIdx.x;
    // Async-stage the rel-pos-bias table into LDS (no conversion needed).
    {
        unsigned base = lds_off(&rpbs[0]);
        for (int i = tid; i < 686; i += 128)
            async_g2l_b128(base + i * 16, rpb + i * 4);
        wait_asynccnt0();
    }
    __syncthreads();

    const int lane = tid & 31;
    const int wave = tid >> 5;
    const int l15  = lane & 15;
    const int half = lane >> 4;
    const int head = blockIdx.x * 4 + wave;

    const int w  = blockIdx.y;
    const int b  = w >> 8;
    const int zb = (w >> 6) & 3;
    const int yb = (w >> 3) & 7;
    const int xb = w & 7;

    // ---- K fragments (B-matrix: K[hd=32, token 16]) ----
    Frag kf[4];
#pragma unroll
    for (int tc = 0; tc < 4; ++tc) {
        int tkn = tc * 16 + l15;
        size_t base = (size_t)tok_pos(b, zb, yb, xb, tkn) * QKV_N + 256 + head * HD;
#pragma unroll
        for (int vg = 0; vg < 8; ++vg) {
            int kk = 2 * vg + 16 * half;
            float2 f = *(const float2*)(qkv + base + kk);
            kf[tc].u[vg] = pk2(f.x, f.y);
        }
    }
    // ---- V fragments (B-matrix: V[token 32 per iter, hd 16]) ----
    Frag vf[2][2];
#pragma unroll
    for (int ki = 0; ki < 2; ++ki)
#pragma unroll
        for (int nb = 0; nb < 2; ++nb) {
            int n = nb * 16 + l15;
#pragma unroll
            for (int vg = 0; vg < 8; ++vg) {
                int t0 = ki * 32 + 16 * half + 2 * vg;
                float e0 = qkv[(size_t)tok_pos(b, zb, yb, xb, t0) * QKV_N + 512 + head * HD + n];
                float e1 = qkv[(size_t)tok_pos(b, zb, yb, xb, t0 + 1) * QKV_N + 512 + head * HD + n];
                vf[ki][nb].u[vg] = pk2(e0, e1);
            }
        }

    const float rs = 0.17677669529663687f;  // 32^-0.5

    // ---- S = Q.K^T + bias, softmax, P -> LDS ----
    for (int tr = 0; tr < 4; ++tr) {
        Frag qf;
        int trow = tr * 16 + l15;
        size_t qbase = (size_t)tok_pos(b, zb, yb, xb, trow) * QKV_N + head * HD;
#pragma unroll
        for (int vg = 0; vg < 8; ++vg) {
            int kk = 2 * vg + 16 * half;
            float2 f = *(const float2*)(qkv + qbase + kk);
            qf.u[vg] = pk2(f.x * rs, f.y * rs);
        }
        v8f s[4] = {};
#pragma unroll
        for (int tc = 0; tc < 4; ++tc)
            s[tc] = wmma_f32_f16(qf.v, kf[tc].v, s[tc]);

        // relative-position bias (computed from coords; rpb staged in LDS)
#pragma unroll
        for (int tc = 0; tc < 4; ++tc)
#pragma unroll
            for (int r = 0; r < 8; ++r) {
                int i = tr * 16 + r + 8 * half;
                int j = tc * 16 + l15;
                int rel = (((i >> 4) - (j >> 4) + 3) * 7 +
                           (((i >> 2) & 3) - ((j >> 2) & 3) + 3)) * 7 +
                          ((i & 3) - (j & 3) + 3);
                s[tc][r] += rpbs[rel * HEADS + head];
            }
        // row softmax: VGPR r holds rows (tr*16+r) / (tr*16+r+8) across lane halves
#pragma unroll
        for (int r = 0; r < 8; ++r) {
            float m = -1e30f;
#pragma unroll
            for (int tc = 0; tc < 4; ++tc) m = fmaxf(m, s[tc][r]);
#pragma unroll
            for (int off = 8; off; off >>= 1) m = fmaxf(m, __shfl_xor(m, off, 32));
            float sum = 0.f;
#pragma unroll
            for (int tc = 0; tc < 4; ++tc) {
                float e = __expf(s[tc][r] - m);
                s[tc][r] = e;
                sum += e;
            }
#pragma unroll
            for (int off = 8; off; off >>= 1) sum += __shfl_xor(sum, off, 32);
            float inv = 1.f / sum;
            int i = tr * 16 + r + 8 * half;
#pragma unroll
            for (int tc = 0; tc < 4; ++tc)
                P[wave][i][tc * 16 + l15] = (_Float16)(s[tc][r] * inv);
        }
    }

    // ---- O = P @ V ----
    for (int tr = 0; tr < 4; ++tr) {
        Frag pa[2];
        int row = tr * 16 + l15;
#pragma unroll
        for (int ki = 0; ki < 2; ++ki)
#pragma unroll
            for (int vg = 0; vg < 8; ++vg) {
                int kk = 2 * vg + 16 * half + 32 * ki;
                pa[ki].u[vg] = *(const unsigned*)&P[wave][row][kk];
            }
#pragma unroll
        for (int nb = 0; nb < 2; ++nb) {
            v8f o = {};
            o = wmma_f32_f16(pa[0].v, vf[0][nb].v, o);
            o = wmma_f32_f16(pa[1].v, vf[1][nb].v, o);
#pragma unroll
            for (int r = 0; r < 8; ++r) {
                int t  = tr * 16 + r + 8 * half;
                int ch = head * HD + nb * 16 + l15;
                attn_out[((size_t)w * 64 + t) * EMB + ch] = o[r];
            }
        }
    }
}

// ---------------------------------------------------------------------------
// Kernel 3: ACmix conv path. Fuses the 1x1 conv (fc_w) into LDS F tiles, then
// grouped 3x3x3 conv. grid (256 tiles, 32 groups, 4 batch), 256 threads.
// Weight tables staged with GLOBAL_LOAD_ASYNC_TO_LDS_B128.
// ---------------------------------------------------------------------------
__global__ __launch_bounds__(256) void acmix_conv(const float* __restrict__ qkv,
                                                  const float* __restrict__ fc_w,
                                                  const float* __restrict__ fc_b,
                                                  const float* __restrict__ dep_w,
                                                  const float* __restrict__ dep_b,
                                                  float* __restrict__ out) {
    __shared__ float Fs[27][240];       // 25920 B
    __shared__ __attribute__((aligned(16))) float Wd[8 * 27 * 27];   // 1458 x b128
    __shared__ __attribute__((aligned(16))) float fw[27 * 24];       //  162 x b128
    __shared__ float fb[27];

    const int tid  = threadIdx.x;
    const int g    = blockIdx.y;   // hd group
    const int b    = blockIdx.z;
    const int tile = blockIdx.x;
    const int z0 = (tile & 7) * 2;
    const int y0 = ((tile >> 3) & 7) * 4;
    const int x0 = (tile >> 6) * 8;

    // Async-stage dep_w slice + fc_w (contiguous f32, no conversion).
    {
        unsigned wdb = lds_off(&Wd[0]);
        const float* src = dep_w + (size_t)g * 5832;
        for (int i = tid; i < 1458; i += 256)
            async_g2l_b128(wdb + i * 16, src + i * 4);
        unsigned fwb = lds_off(&fw[0]);
        for (int i = tid; i < 162; i += 256)
            async_g2l_b128(fwb + i * 16, fc_w + i * 4);
    }
    if (tid < 27) fb[tid] = fc_b[tid];
    wait_asynccnt0();
    __syncthreads();

    // F[b, g, i, voxel] = fc_b[i] + sum_c fc_w[i,c] * qkv_flat[b][c*L*32 + l*32 + g]
    const size_t qb = (size_t)b * (24u * LVOX * 32u);
    for (int v = tid; v < 240; v += 256) {
        int xx = v % 10, yy = (v / 10) % 6, zz = v / 60;
        int z = z0 + zz - 1, y = y0 + yy - 1, x = x0 + xx - 1;
        bool ok = (z >= 0) & (z < 16) & (y >= 0) & (y < 32) & (x >= 0) & (x < 32);
        float qv[24];
        if (ok) {
            int l = (z * 32 + y) * 32 + x;
#pragma unroll
            for (int c = 0; c < 24; ++c)
                qv[c] = qkv[qb + (size_t)c * (LVOX * 32) + (size_t)l * 32 + g];
        } else {
#pragma unroll
            for (int c = 0; c < 24; ++c) qv[c] = 0.f;
        }
#pragma unroll 1
        for (int i = 0; i < 27; ++i) {
            float a = 0.f;
#pragma unroll
            for (int c = 0; c < 24; ++c) a = fmaf(fw[i * 24 + c], qv[c], a);
            Fs[i][v] = ok ? (a + fb[i]) : 0.f;
        }
    }
    __syncthreads();

    // Conv: thread = (oc, lane-strip); each thread does voxels s and s+32.
    const int oc = tid >> 5;
    const int s  = tid & 31;
    const int ly = (s >> 3) & 3;
    const int lx = s & 7;
    float acc0 = 0.f, acc1 = 0.f;
    const float* wrow = &Wd[oc * 729];
#pragma unroll 1
    for (int i = 0; i < 27; ++i) {
        const float* Fi = &Fs[i][0];
#pragma unroll
        for (int dz = 0; dz < 3; ++dz)
#pragma unroll
            for (int dy = 0; dy < 3; ++dy)
#pragma unroll
                for (int dx = 0; dx < 3; ++dx) {
                    float wv = wrow[i * 27 + (dz * 3 + dy) * 3 + dx];
                    int idx = (dz * 6 + (ly + dy)) * 10 + lx + dx;
                    acc0 = fmaf(wv, Fi[idx], acc0);
                    acc1 = fmaf(wv, Fi[idx + 60], acc1);   // z-strip +1
                }
    }
    const int ocg = g * 8 + oc;
    const float bb = dep_b[ocg];
    const int l0 = ((z0 + 0) * 32 + y0 + ly) * 32 + x0 + lx;
    const int l1 = ((z0 + 1) * 32 + y0 + ly) * 32 + x0 + lx;
    out[((size_t)(b * 256 + ocg)) * LVOX + l0] = acc0 + bb;
    out[((size_t)(b * 256 + ocg)) * LVOX + l1] = acc1 + bb;
}

// ---------------------------------------------------------------------------
// Kernel 4: proj GEMM + window reverse + alpha/beta combine (double-buffered).
// ---------------------------------------------------------------------------
__global__ __launch_bounds__(128) void proj_combine(const float* __restrict__ A,
                                                    const float* __restrict__ Bw,
                                                    const float* __restrict__ proj_b,
                                                    const float* __restrict__ conv,
                                                    const float* __restrict__ alpha_p,
                                                    const float* __restrict__ beta_p,
                                                    float* __restrict__ out) {
    __shared__ __attribute__((aligned(16))) _Float16 As[2][64][40];
    __shared__ __attribute__((aligned(16))) _Float16 Bs[2][64][40];

    const int tid  = threadIdx.x;
    const int lane = tid & 31;
    const int wave = tid >> 5;
    const int l15  = lane & 15;
    const int half = lane >> 4;

    const int tileN = blockIdx.x * 64;
    const int tileM = blockIdx.y * 64;
    const int waveM = (wave >> 1) * 32;
    const int waveN = (wave & 1) * 32;

    const float alpha = *alpha_p;
    const float beta  = *beta_p;

    const int bK = (tid >> 4) * 4;
    const int bN = (tid & 15) * 4;

    float4 ra[4], rb[4];
    auto loadT = [&](int k0) {
#pragma unroll
        for (int j = 0; j < 4; ++j) {
            int i = tid + j * 128;
            int row = (i * 4) >> 5, col = (i * 4) & 31;
            ra[j] = *(const float4*)(A + (size_t)(tileM + row) * EMB + k0 + col);
        }
#pragma unroll
        for (int r = 0; r < 4; ++r)
            rb[r] = *(const float4*)(Bw + (size_t)(k0 + bK + r) * EMB + tileN + bN);
    };
    auto stage = [&](int buf) {
#pragma unroll
        for (int j = 0; j < 4; ++j) {
            int i = tid + j * 128;
            int row = (i * 4) >> 5, col = (i * 4) & 31;
            *(uint2*)&As[buf][row][col] =
                make_uint2(pk2(ra[j].x, ra[j].y), pk2(ra[j].z, ra[j].w));
        }
        const float* c0 = &rb[0].x;
        const float* c1 = &rb[1].x;
        const float* c2 = &rb[2].x;
        const float* c3 = &rb[3].x;
#pragma unroll
        for (int nn = 0; nn < 4; ++nn) {
            *(uint2*)&Bs[buf][bN + nn][bK] =
                make_uint2(pk2(c0[nn], c1[nn]), pk2(c2[nn], c3[nn]));
        }
    };

    v8f acc[2][2] = {};
    loadT(0);
    stage(0);
    __syncthreads();

    for (int it = 0; it < 8; ++it) {
        const int buf = it & 1;
        if (it < 7) loadT((it + 1) * 32);

        Frag a[2], b[2];
#pragma unroll
        for (int mb = 0; mb < 2; ++mb) {
            int row = waveM + mb * 16 + l15;
#pragma unroll
            for (int vg = 0; vg < 8; ++vg) {
                int kk = 2 * vg + 16 * half;
                a[mb].u[vg] = *(const unsigned*)&As[buf][row][kk];
            }
        }
#pragma unroll
        for (int nb = 0; nb < 2; ++nb) {
            int n = waveN + nb * 16 + l15;
#pragma unroll
            for (int vg = 0; vg < 8; ++vg) {
                int kk = 2 * vg + 16 * half;
                b[nb].u[vg] = *(const unsigned*)&Bs[buf][n][kk];
            }
        }
#pragma unroll
        for (int mb = 0; mb < 2; ++mb)
#pragma unroll
            for (int nb = 0; nb < 2; ++nb)
                acc[mb][nb] = wmma_f32_f16(a[mb].v, b[nb].v, acc[mb][nb]);

        if (it < 7) {
            stage(1 - buf);
            __syncthreads();
        }
    }

#pragma unroll
    for (int mb = 0; mb < 2; ++mb)
#pragma unroll
        for (int nb = 0; nb < 2; ++nb)
#pragma unroll
            for (int r = 0; r < 8; ++r) {
                int wr  = tileM + waveM + mb * 16 + r + 8 * half; // window-order row
                int col = tileN + waveN + nb * 16 + l15;
                int w = wr >> 6, t = wr & 63;
                int b  = w >> 8, zb = (w >> 6) & 3, yb = (w >> 3) & 7, xb = w & 7;
                int z = zb * 4 + (t >> 4);
                int y = yb * 4 + ((t >> 2) & 3);
                int x = xb * 4 + (t & 3);
                int l = (z * 32 + y) * 32 + x;
                size_t p = (size_t)b * LVOX + l;
                float v = acc[mb][nb][r] + proj_b[col];
                out[p * EMB + col] =
                    alpha * v + beta * conv[((size_t)(b * 256 + col)) * LVOX + l];
            }
}

// ---------------------------------------------------------------------------
extern "C" void kernel_launch(void* const* d_in, const int* in_sizes, int n_in,
                              void* d_out, int out_size, void* d_ws, size_t ws_size,
                              hipStream_t stream) {
    const float* x_in   = (const float*)d_in[0];
    const float* qkv_w  = (const float*)d_in[1];
    const float* proj_w = (const float*)d_in[2];
    const float* proj_b = (const float*)d_in[3];
    const float* rpb    = (const float*)d_in[4];
    const float* fc_w   = (const float*)d_in[5];
    const float* fc_b   = (const float*)d_in[6];
    const float* dep_w  = (const float*)d_in[7];
    const float* dep_b  = (const float*)d_in[8];
    const float* alpha  = (const float*)d_in[9];
    const float* beta   = (const float*)d_in[10];

    float* qkv  = (float*)d_ws;                       // 65536*768  = 50331648 f
    float* attn = qkv  + (size_t)65536 * 768;         // 65536*256  = 16777216 f
    float* conv = attn + (size_t)65536 * 256;         // 4*256*16384= 16777216 f
    float* out  = (float*)d_out;

    qkv_gemm    <<<dim3(12, 1024), 128, 0, stream>>>(x_in, qkv_w, qkv);
    attn_kernel <<<dim3(2, 1024),  128, 0, stream>>>(qkv, rpb, attn);
    acmix_conv  <<<dim3(256, 32, 4), 256, 0, stream>>>(qkv, fc_w, fc_b, dep_w, dep_b, conv);
    proj_combine<<<dim3(4, 1024),  128, 0, stream>>>(attn, proj_w, proj_b, conv,
                                                     alpha, beta, out);
}
